// simple_conv_9603546874032
// MI455X (gfx1250) — matrix operand order repaired and checked by reading, not compiled, not run on hardware
//
#include <hip/hip_runtime.h>
#include <hip/hip_bf16.h>
#include <math.h>

// ---------------------------------------------------------------------------
// Problem constants (from the reference):
//   B=512, H=W=36, N=1296 oscillators, K=8 couplings, episodes=25 (runtime)
//   conv1: 1->8, conv2: 8->16, conv3: 16->32 (all 3x3 SAME over 36x36)
//   fc:    [6144,3456] @ [864,3456]^T + b   -> [512,1296,8] couplings
//   kuramoto: 25 annealed steps, trajectory [26,512,1296]
// Outputs concatenated: phases (26*512*1296 f32) then fc (512*1296*8 f32).
// ---------------------------------------------------------------------------

#define BATCH   512
#define NPIX    1296      // 36*36
#define GEMM_M  6144      // 512*12
#define GEMM_K  3456      // 108*32
#define GEMM_N  864       // 108*8

typedef __attribute__((ext_vector_type(16))) __bf16        v16bf;
typedef __attribute__((ext_vector_type(8)))  float         v8f;
typedef __attribute__((ext_vector_type(4)))  unsigned int  uint4v;
typedef int v4i __attribute__((vector_size(16)));   // matches async-LDS builtin params

union FragU { uint4v q[2]; v16bf v; };

// gfx1250 async global->LDS path (ASYNCcnt), guarded so either toolchain builds.
#if defined(__has_builtin)
#  if __has_builtin(__builtin_amdgcn_global_load_async_to_lds_b128) && \
      __has_builtin(__builtin_amdgcn_s_wait_asynccnt)
#    define USE_ASYNC_LDS 1
#  endif
#endif
#ifndef USE_ASYNC_LDS
#  define USE_ASYNC_LDS 0
#endif

#if USE_ASYNC_LDS
#  define WAIT_ASYNC(n) __builtin_amdgcn_s_wait_asynccnt(n)
#else
#  define WAIT_ASYNC(n) ((void)0)
#endif

static __device__ __forceinline__ void cp16_to_lds(
    unsigned short* dstLds, const unsigned short* srcGlb)
{
#if USE_ASYNC_LDS
    // signature (from hipcc diagnostic): (v4i AS1*, v4i AS3*, imm offset, imm cpol)
    __builtin_amdgcn_global_load_async_to_lds_b128(
        (v4i __attribute__((address_space(1)))*)srcGlb,
        (v4i __attribute__((address_space(3)))*)dstLds,
        /*offset=*/0, /*cpol=*/0);
#else
    *(uint4v*)dstLds = *(const uint4v*)srcGlb;
#endif
}

static __device__ __forceinline__ unsigned short f32_to_bf16_rne(float f) {
    unsigned int u = __float_as_uint(f);
    u += 0x7FFFu + ((u >> 16) & 1u);      // round-to-nearest-even
    return (unsigned short)(u >> 16);
}

// ---------------------------------------------------------------------------
// conv1: [512,1,36,36] -> relu -> [512,8,36,36].  Image + weights in LDS.
// ---------------------------------------------------------------------------
__global__ __launch_bounds__(256) void conv1_kernel(
    const float* __restrict__ in, const float* __restrict__ w,
    const float* __restrict__ bias, float* __restrict__ out)
{
    __shared__ float img[NPIX];
    __shared__ float ws[72];
    __shared__ float bs[8];
    const int b = blockIdx.x, tid = threadIdx.x;
    for (int i = tid; i < NPIX; i += 256) img[i] = in[(size_t)b * NPIX + i];
    if (tid < 72) ws[tid] = w[tid];
    if (tid < 8)  bs[tid] = bias[tid];
    __syncthreads();
    for (int o = tid; o < 8 * NPIX; o += 256) {
        const int c = o / NPIX, p = o - c * NPIX;
        const int y = p / 36, x = p - y * 36;
        float acc = bs[c];
        #pragma unroll
        for (int dy = 0; dy < 3; ++dy) {
            const int yy = y + dy - 1;
            if ((unsigned)yy < 36u) {
                #pragma unroll
                for (int dx = 0; dx < 3; ++dx) {
                    const int xx = x + dx - 1;
                    if ((unsigned)xx < 36u)
                        acc += img[yy * 36 + xx] * ws[c * 9 + dy * 3 + dx];
                }
            }
        }
        out[(size_t)b * (8 * NPIX) + o] = fmaxf(acc, 0.0f);
    }
}

// ---------------------------------------------------------------------------
// conv2: [512,8,36,36] -> relu -> [512,16,36,36].  Full input tile in LDS
// (8*1296*4 = 41.5KB + 4.6KB weights < 64KB).
// ---------------------------------------------------------------------------
__global__ __launch_bounds__(256) void conv2_kernel(
    const float* __restrict__ in, const float* __restrict__ w,
    const float* __restrict__ bias, float* __restrict__ out)
{
    __shared__ float tin[8 * NPIX];
    __shared__ float ws[16 * 8 * 9];
    __shared__ float bs[16];
    const int b = blockIdx.x, tid = threadIdx.x;
    for (int i = tid; i < 8 * NPIX; i += 256) tin[i] = in[(size_t)b * (8 * NPIX) + i];
    for (int i = tid; i < 16 * 8 * 9; i += 256) ws[i] = w[i];
    if (tid < 16) bs[tid] = bias[tid];
    __syncthreads();
    for (int o = tid; o < 16 * NPIX; o += 256) {
        const int co = o / NPIX, p = o - co * NPIX;
        const int y = p / 36, x = p - y * 36;
        float acc = bs[co];
        for (int ci = 0; ci < 8; ++ci) {
            const float* ip = &tin[ci * NPIX];
            const float* wp = &ws[(co * 8 + ci) * 9];
            #pragma unroll
            for (int dy = 0; dy < 3; ++dy) {
                const int yy = y + dy - 1;
                if ((unsigned)yy < 36u) {
                    #pragma unroll
                    for (int dx = 0; dx < 3; ++dx) {
                        const int xx = x + dx - 1;
                        if ((unsigned)xx < 36u)
                            acc += ip[yy * 36 + xx] * wp[dy * 3 + dx];
                    }
                }
            }
        }
        out[(size_t)b * (16 * NPIX) + o] = fmaxf(acc, 0.0f);
    }
}

// ---------------------------------------------------------------------------
// conv3: [512,16,36,36] -> sigmoid -> bf16 GEMM-A [6144,3456].
// Weights (18.4KB) in LDS; input streamed through L1/L2 (83KB/block, L2-resident).
// ---------------------------------------------------------------------------
__global__ __launch_bounds__(256) void conv3_kernel(
    const float* __restrict__ in, const float* __restrict__ w,
    const float* __restrict__ bias, unsigned short* __restrict__ abf)
{
    __shared__ float ws[32 * 16 * 9];
    __shared__ float bs[32];
    const int b = blockIdx.x, tid = threadIdx.x;
    for (int i = tid; i < 32 * 16 * 9; i += 256) ws[i] = w[i];
    if (tid < 32) bs[tid] = bias[tid];
    __syncthreads();
    const float* ibase = in + (size_t)b * (16 * NPIX);
    for (int o = tid; o < 32 * NPIX; o += 256) {
        const int co = o / NPIX, p = o - co * NPIX;
        const int y = p / 36, x = p - y * 36;
        float acc = bs[co];
        for (int ci = 0; ci < 16; ++ci) {
            const float* ip = ibase + ci * NPIX;
            const float* wp = &ws[(co * 16 + ci) * 9];
            #pragma unroll
            for (int dy = 0; dy < 3; ++dy) {
                const int yy = y + dy - 1;
                if ((unsigned)yy < 36u) {
                    #pragma unroll
                    for (int dx = 0; dx < 3; ++dx) {
                        const int xx = x + dx - 1;
                        if ((unsigned)xx < 36u)
                            acc += ip[yy * 36 + xx] * wp[dy * 3 + dx];
                    }
                }
            }
        }
        const float s = 1.0f / (1.0f + __expf(-acc));
        abf[(size_t)b * (32 * NPIX) + o] = f32_to_bf16_rne(s);
    }
}

// ---------------------------------------------------------------------------
// fc_w f32 -> bf16 (once per launch; it's a kernel input, no caching allowed)
// ---------------------------------------------------------------------------
__global__ __launch_bounds__(256) void cvt_bf16_kernel(
    const float* __restrict__ src, unsigned short* __restrict__ dst, int n)
{
    const int i = blockIdx.x * 256 + threadIdx.x;
    if (i < n) dst[i] = f32_to_bf16_rne(src[i]);
}

// ---------------------------------------------------------------------------
// WMMA GEMM: C[6144,864] = A[6144,3456](bf16) * W[864,3456]^T(bf16) + bias.
// Block tile 128x96, BK=32, 256 threads = 8 waves; wave w owns rows
// [16w,16w+16) x all 96 cols -> 6 accumulators, 6 back-to-back v_wmma/K-step.
// Triple-buffered LDS (42KB) fed by async global->LDS copies (ASYNCcnt):
// tile k+2 is issued right after the barrier of step k, so loads have two
// full K-steps of latency cover and only ONE barrier per K-step is needed
// (a buffer written at step k was last read at step k-1, protected by the
// barrier of step k).  Per wave each tile is exactly 4 async ops
// => s_wait_asynccnt 4 == current tile resident, next tile still in flight.
// ---------------------------------------------------------------------------
#define BM 128
#define BN 96
#define BK 32
#define KT (GEMM_K / BK)   // 108

__global__ __launch_bounds__(256) void fc_gemm_wmma_kernel(
    const unsigned short* __restrict__ A,   // [6144][3456] bf16 bits
    const unsigned short* __restrict__ W,   // [864][3456]  bf16 bits
    const float* __restrict__ bias,         // [864]
    float* __restrict__ C)                  // [6144][864]
{
    __shared__ __align__(16) unsigned short As[3][BM][BK];  // 3 x 8 KB
    __shared__ __align__(16) unsigned short Bs[3][BN][BK];  // 3 x 6 KB

    const int tid  = threadIdx.x;
    const int wave = tid >> 5;          // wave32
    const int lane = tid & 31;
    const int lm   = lane & 15;         // position within 16-lane half
    const int lh   = lane >> 4;         // lane half (0/1)
    const int m0   = blockIdx.x * BM;
    const int n0   = blockIdx.y * BN;

    // issue one K-tile (128x32 A + 96x32 B) into LDS buffer bi
    auto issue_tile = [&](int k0, int bi) {
        int s = tid, r = s >> 2, q = s & 3;
        cp16_to_lds(&As[bi][r][q * 8], A + (size_t)(m0 + r) * GEMM_K + k0 + q * 8);
        s = tid + 256; r = s >> 2; q = s & 3;
        cp16_to_lds(&As[bi][r][q * 8], A + (size_t)(m0 + r) * GEMM_K + k0 + q * 8);
        s = tid; r = s >> 2; q = s & 3;
        cp16_to_lds(&Bs[bi][r][q * 8], W + (size_t)(n0 + r) * GEMM_K + k0 + q * 8);
        if (tid < 128) {
            s = tid + 256; r = s >> 2; q = s & 3;
            cp16_to_lds(&Bs[bi][r][q * 8], W + (size_t)(n0 + r) * GEMM_K + k0 + q * 8);
        } else {
            // keep async-op count uniform across waves (deterministic ASYNCcnt)
            s = tid; r = s >> 2; q = s & 3;
            cp16_to_lds(&Bs[bi][r][q * 8], W + (size_t)(n0 + r) * GEMM_K + k0 + q * 8);
        }
    };

    v8f acc[6];
    #pragma unroll
    for (int i = 0; i < 6; ++i) acc[i] = (v8f){0,0,0,0,0,0,0,0};

    const int am = wave * 16 + lm;

    issue_tile(0,  0);      // prologue: tiles 0 and 1 in flight
    issue_tile(BK, 1);

    int cur = 0;
    for (int kt = 0; kt < KT; ++kt) {
        if (kt + 1 < KT) { WAIT_ASYNC(4); } else { WAIT_ASYNC(0); }
        __syncthreads();                     // single barrier per K-step
        if (kt + 2 < KT) {
            int wb = cur + 2; if (wb >= 3) wb -= 3;
            issue_tile((kt + 2) * BK, wb);   // buffer last read @ kt-1, barrier-protected
        }

        // --- A fragment: row m, K in {8h..8h+7} U {16+8h..16+8h+7} ---
        FragU afr;
        afr.q[0] = *(const uint4v*)&As[cur][am][8 * lh];
        afr.q[1] = *(const uint4v*)&As[cur][am][16 + 8 * lh];

        // --- all six B fragments first, then back-to-back WMMAs ---
        FragU bfr[6];
        #pragma unroll
        for (int nt = 0; nt < 6; ++nt) {
            const int bn = nt * 16 + lm;
            bfr[nt].q[0] = *(const uint4v*)&Bs[cur][bn][8 * lh];
            bfr[nt].q[1] = *(const uint4v*)&Bs[cur][bn][16 + 8 * lh];
        }
        #pragma unroll
        for (int nt = 0; nt < 6; ++nt)
            acc[nt] = __builtin_amdgcn_wmma_f32_16x16x32_bf16(
                false, afr.v, false, bfr[nt].v, (short)0, acc[nt], false, false);

        cur = (cur + 1 == 3) ? 0 : cur + 1;
    }

    // --- store D + bias: VGPR r -> M = r + 8*lh, N = lm ---
    #pragma unroll
    for (int nt = 0; nt < 6; ++nt) {
        const int col = n0 + nt * 16 + lm;
        const float bv = bias[col];
        #pragma unroll
        for (int r = 0; r < 8; ++r) {
            const int row = m0 + wave * 16 + r + 8 * lh;
            C[(size_t)row * GEMM_N + col] = acc[nt][r] + bv;
        }
    }
}

// ---------------------------------------------------------------------------
// Kuramoto: one block per batch. Phases live in LDS (5.2KB); each thread's
// connectivity (48 ints) + couplings (48 floats) hoisted into registers.
//   delta_i = ep * sum_j c_ij * sin(phase[nbr_ij] - phase_i)
//   ep_{s+1} = ep_s * (1 - 0.5*s/steps)           (delta uses pre-update ep)
// Writes trajectory snapshots t = 0..steps.
// ---------------------------------------------------------------------------
__global__ __launch_bounds__(256) void kuramoto_kernel(
    const float* __restrict__ phase0,     // [512][1296]
    const int*   __restrict__ conn,       // [512][1296][8]
    const float* __restrict__ coupling,   // [512][1296][8] (fc region of d_out)
    const int*   __restrict__ episodes_p,
    float* __restrict__ out)              // [steps+1][512][1296]
{
    __shared__ float ph[NPIX];
    const int b = blockIdx.x, tid = threadIdx.x;
    const int steps = *episodes_p;

    for (int n = tid; n < NPIX; n += 256) {
        const float p = phase0[(size_t)b * NPIX + n];
        ph[n] = p;
        out[(size_t)b * NPIX + n] = p;     // t = 0 snapshot
    }

    int   cidx[6][8];
    float cw[6][8];
    #pragma unroll
    for (int it = 0; it < 6; ++it) {
        const int n = tid + it * 256;
        const int nn = (n < NPIX) ? n : 0;
        const size_t base = ((size_t)b * NPIX + nn) * 8;
        #pragma unroll
        for (int j = 0; j < 8; ++j) {
            cidx[it][j] = conn[base + j];
            cw[it][j]   = coupling[base + j];
        }
    }
    __syncthreads();

    float ep = 0.1f;
    const float inv_steps = 1.0f / (float)steps;
    for (int s = 0; s < steps; ++s) {
        float np[6];
        #pragma unroll
        for (int it = 0; it < 6; ++it) {
            const int n = tid + it * 256;
            if (n < NPIX) {
                const float my = ph[n];
                float sum = 0.0f;
                #pragma unroll
                for (int j = 0; j < 8; ++j)
                    sum += cw[it][j] * __sinf(ph[cidx[it][j]] - my);
                np[it] = my + ep * sum;
            }
        }
        __syncthreads();   // all gathers done before anyone writes
        float* o = out + (size_t)(s + 1) * BATCH * NPIX + (size_t)b * NPIX;
        #pragma unroll
        for (int it = 0; it < 6; ++it) {
            const int n = tid + it * 256;
            if (n < NPIX) { ph[n] = np[it]; o[n] = np[it]; }
        }
        __syncthreads();   // writes visible before next step's gathers
        ep = ep - 0.5f * (float)s * ep * inv_steps;
    }
}

// ---------------------------------------------------------------------------
// Launch: conv1 -> conv2 -> conv3(bf16 A) ; fc_w->bf16 ; WMMA GEMM (+bias)
// -> kuramoto.  All on `stream`, workspace-only scratch (~112 MB needed).
// ---------------------------------------------------------------------------
extern "C" void kernel_launch(void* const* d_in, const int* in_sizes, int n_in,
                              void* d_out, int out_size, void* d_ws, size_t ws_size,
                              hipStream_t stream) {
    const float* input   = (const float*)d_in[0];
    const float* phase0  = (const float*)d_in[1];
    const int*   conn    = (const int*)  d_in[2];
    const int*   episodes= (const int*)  d_in[3];
    const float* w1      = (const float*)d_in[4];
    const float* b1      = (const float*)d_in[5];
    const float* w2      = (const float*)d_in[6];
    const float* b2      = (const float*)d_in[7];
    const float* w3      = (const float*)d_in[8];
    const float* b3      = (const float*)d_in[9];
    const float* fcw     = (const float*)d_in[10];
    const float* fcb     = (const float*)d_in[11];

    // workspace layout (all offsets 256B-aligned)
    char* ws = (char*)d_ws;
    float*          c1  = (float*)(ws);                                  // 21,233,664 B
    float*          c2  = (float*)(ws + 21233664);                       // 42,467,328 B
    unsigned short* Abf = (unsigned short*)(ws + 21233664 + 42467328);   // 42,467,328 B
    unsigned short* Wbf = (unsigned short*)(ws + 21233664 + 42467328 + 42467328); // 5,971,968 B

    float* out   = (float*)d_out;
    float* fcOut = out + ((size_t)out_size - (size_t)BATCH * NPIX * 8);  // fc region

    conv1_kernel<<<BATCH, 256, 0, stream>>>(input, w1, b1, c1);
    conv2_kernel<<<BATCH, 256, 0, stream>>>(c1, w2, b2, c2);
    conv3_kernel<<<BATCH, 256, 0, stream>>>(c2, w3, b3, Abf);

    const int wn = GEMM_N * GEMM_K;  // 2,985,984
    cvt_bf16_kernel<<<(wn + 255) / 256, 256, 0, stream>>>(fcw, Wbf, wn);

    dim3 ggrid(GEMM_M / BM, GEMM_N / BN);  // 48 x 9
    fc_gemm_wmma_kernel<<<ggrid, 256, 0, stream>>>(Abf, Wbf, fcb, fcOut);

    kuramoto_kernel<<<BATCH, 256, 0, stream>>>(phase0, conn, fcOut, episodes, out);
}
